// MultiHeadSelfAttention_36129264894565
// MI455X (gfx1250) — matrix operand order repaired and checked by reading, not compiled, not run on hardware
//
#include <hip/hip_runtime.h>
#include <hip/hip_bf16.h>

// MHA for MI455X (gfx1250, wave32, WMMA 16x16x32 f16 -> f32 acc).
// B=2, S=1024, D=1024, H=16, DH=64.

typedef __attribute__((ext_vector_type(16))) _Float16 v16h;
typedef __attribute__((ext_vector_type(8)))  _Float16 v8h;
typedef __attribute__((ext_vector_type(8)))  float    v8f;

#define NB   2
#define NS   1024
#define ND   1024
#define NH   16
#define NDH  64
#define NROW (NB * NS)      // 2048  (flattened B*S rows)
#define N3D  (3 * ND)       // 3072

// ---------------------------------------------------------------------------
// WMMA helpers
// ---------------------------------------------------------------------------
__device__ __forceinline__ v8f wmma16x16x32(v16h a, v16h b, v8f c) {
  // (neg_a, A, neg_b, B, c_mod, C, reuse_a, reuse_b)
  return __builtin_amdgcn_wmma_f32_16x16x32_f16(false, a, false, b, (short)0, c,
                                                false, false);
}

// A-fragment (16x32, f16).  Lanes 0-15 hold M=0..15 with
// K = 0..15; lanes 16-31 hold the same rows with K = 16..31 (matches B).
// `tile` points at (row 0, k 0) of the 16x32 tile; `ld` = row stride in halves.
__device__ __forceinline__ v16h load_a_frag(const _Float16* tile, int lane, int ld) {
  const int m  = lane & 15;
  const int hi = lane >> 4;                 // 0 or 1
  const _Float16* p = tile + m * ld;
  v8h lo = *(const v8h*)(p + (hi ? 16 : 0));
  v8h hh = *(const v8h*)(p + (hi ? 24 : 8));
  v16h r;
#pragma unroll
  for (int i = 0; i < 8; ++i) { r[i] = lo[i]; r[i + 8] = hh[i]; }
  return r;
}

// B-fragment (32x16, f16).  Lane n%16 = column; 16 halves = contiguous K
// range starting at 16*(lane/16).  `tile` points at (k 0, col 0) stored
// column-major-in-K (i.e. each column's K values contiguous, stride `ld`).
__device__ __forceinline__ v16h load_b_frag(const _Float16* tile, int lane, int ld) {
  const int n = lane & 15;
  const int k = (lane >> 4) << 4;           // 0 or 16
  return *(const v16h*)(tile + n * ld + k);
}

__device__ __forceinline__ float hmax16(float v) {
#pragma unroll
  for (int m = 1; m <= 8; m <<= 1) v = fmaxf(v, __shfl_xor(v, m));
  return v;          // masks < 16: stays within each 16-lane half of the wave
}
__device__ __forceinline__ float hsum16(float v) {
#pragma unroll
  for (int m = 1; m <= 8; m <<= 1) v += __shfl_xor(v, m);
  return v;
}

// ---------------------------------------------------------------------------
// 32x64 wave tile GEMM core.  NOTE: deliberately NOT manually double-buffered:
// with 64 accumulator VGPRs + 6 fragment registers, a second fragment set
// spills accumulators to scratch (observed in round 2).  The compiler's own
// 4-step unroll + clause-batched loads is the best schedule at this tile size.
// ---------------------------------------------------------------------------
__device__ __forceinline__ void gemm_core_32x64(const _Float16* __restrict__ A,
                                                const _Float16* __restrict__ Bm,
                                                int lane, v8f acc[2][4]) {
  for (int k = 0; k < ND; k += 32) {
    v16h a0 = load_a_frag(A + k, lane, ND);
    v16h a1 = load_a_frag(A + 16 * ND + k, lane, ND);
#pragma unroll
    for (int j = 0; j < 4; ++j) {
      v16h bm = load_b_frag(Bm + j * 16 * ND + k, lane, ND);
      acc[0][j] = wmma16x16x32(a0, bm, acc[0][j]);
      acc[1][j] = wmma16x16x32(a1, bm, acc[1][j]);
    }
  }
}

// ---------------------------------------------------------------------------
// Kernel 0: f32 -> f16 convert
// ---------------------------------------------------------------------------
__global__ void k_f32_to_f16(const float* __restrict__ src,
                             _Float16* __restrict__ dst, int n) {
  int i = blockIdx.x * blockDim.x + threadIdx.x;
  int stride = gridDim.x * blockDim.x;
  for (; i < n; i += stride) dst[i] = (_Float16)src[i];
}

// ---------------------------------------------------------------------------
// Kernel 1: QKV projection.  X[2048,1024] x Wqkv^T -> Q/K [B,H,S,DH],
// V transposed [B,H,DH,S].  Wave tile 32x64, 8 waves/block.
// ---------------------------------------------------------------------------
__global__ void k_qkv_gemm(const _Float16* __restrict__ X,
                           const _Float16* __restrict__ W,
                           _Float16* __restrict__ Q,
                           _Float16* __restrict__ Kc,
                           _Float16* __restrict__ Vt) {
  const int lane = threadIdx.x & 31;
  const int wid  = blockIdx.x * (blockDim.x >> 5) + (threadIdx.x >> 5);
  const int mt = wid % (NROW / 32);         // 64 m-tiles
  const int nt = wid / (NROW / 32);         // 48 n-tiles
  const int m0 = mt * 32, n0 = nt * 64;

  v8f acc[2][4];
#pragma unroll
  for (int i = 0; i < 2; ++i)
#pragma unroll
    for (int j = 0; j < 4; ++j)
#pragma unroll
      for (int r = 0; r < 8; ++r) acc[i][j][r] = 0.0f;

  gemm_core_32x64(X + (size_t)m0 * ND, W + (size_t)n0 * ND, lane, acc);

  // scatter epilogue: C/D layout -> vgpr r = row (lanes 16-31: +8), lane%16 = col
#pragma unroll
  for (int i = 0; i < 2; ++i)
#pragma unroll
    for (int j = 0; j < 4; ++j)
#pragma unroll
      for (int r = 0; r < 8; ++r) {
        const int m = m0 + 16 * i + r + ((lane >> 4) << 3);
        const int e = n0 + 16 * j + (lane & 15);
        const float val = acc[i][j][r];
        const int which = e >> 10;          // 0=q 1=k 2=v
        const int d  = e & (ND - 1);
        const int h  = d >> 6;
        const int dh = d & (NDH - 1);
        const int b  = m >> 10;
        const int s  = m & (NS - 1);
        const int bh = b * NH + h;
        if (which == 0)
          Q[((size_t)bh * NS + s) * NDH + dh] = (_Float16)val;
        else if (which == 1)
          Kc[((size_t)bh * NS + s) * NDH + dh] = (_Float16)val;
        else
          Vt[((size_t)bh * NDH + dh) * NS + s] = (_Float16)val;
      }
}

// ---------------------------------------------------------------------------
// Kernel 2: flash attention.  One wave per (b, h, 16-row q tile).
// scores = Q K^T * 0.125, online softmax, O += P V.  P re-swizzled via LDS.
// V B-fragment loads are hoisted ahead of the softmax VALU block so they
// are in flight behind the exp/shuffle work (only 4 accumulators here, so
// the extra 32 fragment VGPRs fit without spilling).
// ---------------------------------------------------------------------------
__global__ void k_attention(const _Float16* __restrict__ Q,
                            const _Float16* __restrict__ Kc,
                            const _Float16* __restrict__ Vt,
                            _Float16* __restrict__ Ctx) {
  __shared__ _Float16 lds[8 * 16 * 40];     // 8 waves x (16 rows x 32 cols, pad 8)
  const int lane = threadIdx.x & 31;
  const int widx = threadIdx.x >> 5;
  const int wid  = blockIdx.x * 8 + widx;
  const int qt = wid & 63;                  // S/16 q tiles
  const int h  = (wid >> 6) & (NH - 1);
  const int b  = wid >> 10;
  const int bh = b * NH + h;
  const float scale = 0.125f;               // 1/sqrt(64)

  const _Float16* Qb = Q  + ((size_t)bh * NS + qt * 16) * NDH;
  const _Float16* Kb = Kc + (size_t)bh * NS * NDH;
  const _Float16* Vb = Vt + (size_t)bh * NDH * NS;
  _Float16* lp = &lds[widx * 640];

  const v16h aq0 = load_a_frag(Qb + 0,  lane, NDH);
  const v16h aq1 = load_a_frag(Qb + 32, lane, NDH);

  v8f acc[4];
  float mr[8], lr[8];
#pragma unroll
  for (int j = 0; j < 4; ++j)
#pragma unroll
    for (int r = 0; r < 8; ++r) acc[j][r] = 0.0f;
#pragma unroll
  for (int r = 0; r < 8; ++r) { mr[r] = -3.0e38f; lr[r] = 0.0f; }

  for (int kt = 0; kt < NS; kt += 32) {
    // ---- scores: two 16x16 tiles (keys kt..kt+15, kt+16..kt+31) ----
    v8f sc[2];
#pragma unroll
    for (int sub = 0; sub < 2; ++sub) {
      const _Float16* kbase = Kb + (size_t)(kt + 16 * sub) * NDH;
      v8f s;
#pragma unroll
      for (int r = 0; r < 8; ++r) s[r] = 0.0f;
      s = wmma16x16x32(aq0, load_b_frag(kbase + 0,  lane, NDH), s);
      s = wmma16x16x32(aq1, load_b_frag(kbase + 32, lane, NDH), s);
      sc[sub] = s;
    }
    // ---- issue V fragment loads now; consumed after softmax ----
    v16h bv[4];
#pragma unroll
    for (int j = 0; j < 4; ++j)
      bv[j] = load_b_frag(Vb + (size_t)(j * 16) * NS + kt, lane, NS);
    // ---- online softmax (row stats per vgpr row, shared across 16 lanes) ----
#pragma unroll
    for (int r = 0; r < 8; ++r) {
      const float s0 = sc[0][r] * scale;
      const float s1 = sc[1][r] * scale;
      const float tmax = hmax16(fmaxf(s0, s1));
      const float mnew = fmaxf(mr[r], tmax);
      const float corr = __expf(mr[r] - mnew);
      const float p0 = __expf(s0 - mnew);
      const float p1 = __expf(s1 - mnew);
      const float rs = hsum16(p0 + p1);
      lr[r] = lr[r] * corr + rs;
      mr[r] = mnew;
#pragma unroll
      for (int j = 0; j < 4; ++j) acc[j][r] *= corr;
      const int m = r + ((lane >> 4) << 3);
      lp[m * 40 + (lane & 15)]      = (_Float16)p0;
      lp[m * 40 + 16 + (lane & 15)] = (_Float16)p1;
    }
    asm volatile("s_wait_dscnt 0" ::: "memory");
    // ---- P (16x32) x V (32x64) ----
    const v16h ap = load_a_frag(lp, lane, 40);
#pragma unroll
    for (int j = 0; j < 4; ++j) acc[j] = wmma16x16x32(ap, bv[j], acc[j]);
  }

  // ---- normalize + write context as [B,S,D] (f16) ----
#pragma unroll
  for (int r = 0; r < 8; ++r) {
    const float inv = 1.0f / lr[r];
    const int m = r + ((lane >> 4) << 3);
    const int s = qt * 16 + m;
#pragma unroll
    for (int j = 0; j < 4; ++j) {
      const int dcol = h * NDH + j * 16 + (lane & 15);
      Ctx[((size_t)(b * NS + s)) * ND + dcol] = (_Float16)(acc[j][r] * inv);
    }
  }
}

// ---------------------------------------------------------------------------
// Kernel 3: output projection.  Ctx[2048,1024] x Wout^T -> out f32 [B,S,D].
// ---------------------------------------------------------------------------
__global__ void k_out_gemm(const _Float16* __restrict__ Ctx,
                           const _Float16* __restrict__ W,
                           float* __restrict__ Out) {
  const int lane = threadIdx.x & 31;
  const int wid  = blockIdx.x * (blockDim.x >> 5) + (threadIdx.x >> 5);
  const int mt = wid % (NROW / 32);         // 64 m-tiles
  const int nt = wid / (NROW / 32);         // 16 n-tiles
  const int m0 = mt * 32, n0 = nt * 64;

  v8f acc[2][4];
#pragma unroll
  for (int i = 0; i < 2; ++i)
#pragma unroll
    for (int j = 0; j < 4; ++j)
#pragma unroll
      for (int r = 0; r < 8; ++r) acc[i][j][r] = 0.0f;

  gemm_core_32x64(Ctx + (size_t)m0 * ND, W + (size_t)n0 * ND, lane, acc);

#pragma unroll
  for (int i = 0; i < 2; ++i)
#pragma unroll
    for (int j = 0; j < 4; ++j)
#pragma unroll
      for (int r = 0; r < 8; ++r) {
        const int m = m0 + 16 * i + r + ((lane >> 4) << 3);
        const int n = n0 + 16 * j + (lane & 15);
        Out[(size_t)m * ND + n] = acc[i][j][r];
      }
}

// ---------------------------------------------------------------------------
// Launch
// ---------------------------------------------------------------------------
extern "C" void kernel_launch(void* const* d_in, const int* in_sizes, int n_in,
                              void* d_out, int out_size, void* d_ws, size_t ws_size,
                              hipStream_t stream) {
  const float* x    = (const float*)d_in[0];
  const float* wqkv = (const float*)d_in[1];
  const float* wout = (const float*)d_in[2];
  float* out = (float*)d_out;

  _Float16* ws = (_Float16*)d_ws;
  size_t off = 0;
  _Float16* x16  = ws + off; off += (size_t)NROW * ND;        // 2M
  _Float16* wq16 = ws + off; off += (size_t)N3D * ND;         // 3M
  _Float16* wo16 = ws + off; off += (size_t)ND * ND;          // 1M
  _Float16* q16  = ws + off; off += (size_t)NB * NH * NS * NDH;
  _Float16* k16  = ws + off; off += (size_t)NB * NH * NS * NDH;
  _Float16* vt16 = ws + off; off += (size_t)NB * NH * NDH * NS;
  _Float16* ctx16 = ws + off; off += (size_t)NROW * ND;
  (void)ws_size; (void)in_sizes; (void)n_in; (void)out_size;

  k_f32_to_f16<<<2048, 256, 0, stream>>>(x, x16, NROW * ND);
  k_f32_to_f16<<<2048, 256, 0, stream>>>(wqkv, wq16, N3D * ND);
  k_f32_to_f16<<<1024, 256, 0, stream>>>(wout, wo16, ND * ND);

  // 64 m-tiles * 48 n-tiles = 3072 waves, 8 waves/block
  k_qkv_gemm<<<384, 256, 0, stream>>>(x16, wq16, q16, k16, vt16);
  // 2 * 16 * 64 = 2048 waves
  k_attention<<<256, 256, 0, stream>>>(q16, k16, vt16, ctx16);
  // 64 * 16 = 1024 waves
  k_out_gemm<<<128, 256, 0, stream>>>(ctx16, wo16, out);
}
